// Gpt2Attention_29953101923104
// MI455X (gfx1250) — matrix operand order repaired and checked by reading, not compiled, not run on hardware
//
#include <hip/hip_runtime.h>

// ---------------------------------------------------------------------------
// GPT-2 attention layer on gfx1250 (CDNA5, wave32, WMMA bf16 + TDM).
// Pipeline: f32->bf16 convert, QKV GEMM (TDM-fed LDS + WMMA), flash attention
// (WMMA + online softmax), output projection (TDM-fed LDS + WMMA).
// ---------------------------------------------------------------------------

typedef __attribute__((ext_vector_type(16))) __bf16 v16bf;
typedef __attribute__((ext_vector_type(8)))  float  v8f;
typedef __attribute__((ext_vector_type(4)))  unsigned int v4u;
typedef __attribute__((ext_vector_type(8)))  int v8i;
typedef __attribute__((ext_vector_type(4)))  int v4i;

#define LOG2E 1.44269504088896340736f

// Problem constants (from reference).
#define BB   4
#define SS   2048
#define DD   1024
#define HH   16
#define HD   64
#define N3   (3 * DD)   // 3072
#define BH   (BB * HH)  // 64

__device__ __forceinline__ unsigned short f2bf(float f) {
  unsigned int u = __builtin_bit_cast(unsigned int, f);
  u += 0x7FFFu + ((u >> 16) & 1u);   // round-to-nearest-even
  return (unsigned short)(u >> 16);
}

struct Frag {
  union { v16bf v; unsigned int u[8]; };
};

// A-fragment (16x32 bf16, M x K): lane holds row M = lane&15.
// VGPR v: K = (v/4)*16 + (lane/16)*8 + (v%4)*2, packed pair. (ISA 7.12.2)
__device__ __forceinline__ Frag load_a_frag(const unsigned short* __restrict__ src,
                                            int ld, int lane) {
  Frag f;
  const int m = lane & 15, half = lane >> 4;
  const unsigned short* row = src + (size_t)m * ld;
#pragma unroll
  for (int v = 0; v < 8; ++v) {
    const int kk = ((v >> 2) << 4) + (half << 3) + ((v & 3) << 1);
    f.u[v] = *(const unsigned int*)(row + kk);
  }
  return f;
}

// B-fragment (32x16 bf16, K x N): lane holds col N = lane&15.
// VGPR v: K = (v/4)*16 + (lane/16)*8 + (v%4)*2, packed pair. src is N-major with K contiguous.
__device__ __forceinline__ Frag load_b_frag(const unsigned short* __restrict__ src,
                                            int ld, int lane) {
  Frag f;
  const int n = lane & 15, half = lane >> 4;
  const unsigned short* row = src + (size_t)n * ld;
#pragma unroll
  for (int v = 0; v < 8; ++v) {
    const int kk = ((v >> 2) << 4) + (half << 3) + ((v & 3) << 1);
    f.u[v] = *(const unsigned int*)(row + kk);
  }
  return f;
}

__device__ __forceinline__ v8f wmma_bf16(const Frag& a, const Frag& b, v8f c) {
  return __builtin_amdgcn_wmma_f32_16x16x32_bf16(false, a.v, false, b.v,
                                                 (short)0, c, false, false);
}

// ---------------------------------------------------------------------------
// Tensor Data Mover: 2-D tile load (bf16 elements) Global -> LDS.
// D# layout per CDNA5 ISA ch.8: group0 = {count/flags, lds_addr,
// global_addr[56:0], type=2}; group1 = {data_size, tensor dims, tile dims,
// dim0 stride}. Groups 2/3 zero (<=2D tensor). 6-arg builtin form (clang-23).
__device__ __forceinline__ void tdm_load_2d(unsigned lds_off,
                                            unsigned long long gaddr,
                                            unsigned tensor_d0, unsigned tensor_d1,
                                            unsigned tile_d0, unsigned tile_d1,
                                            unsigned stride0_elems) {
  v4u g0;
  g0.x = 1u;                                               // count=1 (valid)
  g0.y = lds_off;                                          // lds_addr (bytes)
  g0.z = (unsigned)(gaddr & 0xFFFFFFFFull);                // global_addr[31:0]
  g0.w = (unsigned)((gaddr >> 32) & 0x1FFFFFFull) | (2u << 30); // [56:32]|type=2
  v8i g1;
  g1[0] = (int)(1u << 16);                                 // data_size=1 -> 2B
  g1[1] = (int)((tensor_d0 & 0xFFFFu) << 16);              // tensor_dim0 lo
  g1[2] = (int)(((tensor_d0 >> 16) & 0xFFFFu) |
                ((tensor_d1 & 0xFFFFu) << 16));            // dim0 hi | dim1 lo
  g1[3] = (int)(((tensor_d1 >> 16) & 0xFFFFu) |
                ((tile_d0 & 0xFFFFu) << 16));              // dim1 hi | tile_dim0
  g1[4] = (int)(tile_d1 & 0xFFFFu);                        // tile_dim1 (dim2=0)
  g1[5] = (int)stride0_elems;                              // dim0_stride[31:0]
  g1[6] = 0;                                               // stride hi bits
  g1[7] = 0;
  v4i z4 = {0, 0, 0, 0};
  v8i z8 = {0, 0, 0, 0, 0, 0, 0, 0};
  __builtin_amdgcn_tensor_load_to_lds(g0, g1, z4, z4, z8, 0);
}

// ---------------------------------------------------------------------------
// f32 -> bf16 elementwise convert (n multiple of 4).
__global__ void cvt_bf16_kernel(const float* __restrict__ in,
                                unsigned short* __restrict__ out, int n) {
  const int i = (blockIdx.x * blockDim.x + threadIdx.x) * 4;
  if (i + 3 < n) {
    float4 f = *(const float4*)(in + i);
    out[i + 0] = f2bf(f.x);
    out[i + 1] = f2bf(f.y);
    out[i + 2] = f2bf(f.z);
    out[i + 3] = f2bf(f.w);
  }
}

// in: [K, N] f32 row-major -> out: [N, K] bf16 (so WMMA B-frags read contiguous K)
__global__ void transpose_cvt_kernel(const float* __restrict__ in,
                                     unsigned short* __restrict__ out,
                                     int K, int N) {
  const int t = blockIdx.x * blockDim.x + threadIdx.x;
  if (t >= K * N) return;
  const int n = t / K, k = t - n * K;
  out[t] = f2bf(in[(size_t)k * N + n]);
}

// ---------------------------------------------------------------------------
// QKV GEMM: [8192,1024]bf16 @ [1024,3072] + bias. Block = 128x64 tile, 8 waves.
// Weight tile (64 N x 32 K) double-buffered in LDS via TDM, shared by 8 waves.
// Scatters into Q,K [BH,S,64] and V transposed [BH,64,S], all bf16.
__global__ void qkv_gemm_kernel(const unsigned short* __restrict__ xb,
                                const unsigned short* __restrict__ wT,
                                const float* __restrict__ bias,
                                unsigned short* __restrict__ Qb,
                                unsigned short* __restrict__ Kb,
                                unsigned short* __restrict__ Vt) {
  __shared__ unsigned short Btile[2][64 * 32];   // 2 x 4 KB
  const int lane = threadIdx.x & 31;
  const int wave = threadIdx.x >> 5;
  const int tilesN = N3 / 64;                    // 48
  const int m0 = (blockIdx.x / tilesN) * 128 + wave * 16;
  const int n0 = (blockIdx.x % tilesN) * 64;

  const unsigned long long wbase = (unsigned long long)(const void*)wT;
  const unsigned ldsb0 = (unsigned)(size_t)&Btile[0][0];
  const unsigned ldsb1 = (unsigned)(size_t)&Btile[1][0];

  if (wave == 0) {
    tdm_load_2d(ldsb0, wbase + (size_t)n0 * DD * 2, DD, N3, 32, 64, DD);
    __builtin_amdgcn_s_wait_tensorcnt(0);
  }
  __syncthreads();

  v8f c[4] = {};
  int cur = 0;
  for (int k0 = 0; k0 < DD; k0 += 32) {
    if (wave == 0 && k0 + 32 < DD)
      tdm_load_2d(cur ? ldsb0 : ldsb1,
                  wbase + ((size_t)n0 * DD + k0 + 32) * 2, DD, N3, 32, 64, DD);

    const Frag a = load_a_frag(xb + (size_t)m0 * DD + k0, DD, lane);
#pragma unroll
    for (int nt = 0; nt < 4; ++nt) {
      const Frag b = load_b_frag(&Btile[cur][nt * 16 * 32], 32, lane);
      c[nt] = wmma_bf16(a, b, c[nt]);
    }

    if (wave == 0 && k0 + 32 < DD) __builtin_amdgcn_s_wait_tensorcnt(0);
    __syncthreads();
    cur ^= 1;
  }

  const int half = lane >> 4, nc = lane & 15;
#pragma unroll
  for (int nt = 0; nt < 4; ++nt) {
#pragma unroll
    for (int v = 0; v < 8; ++v) {
      const int n = n0 + nt * 16 + nc;
      const int m = m0 + v + 8 * half;
      const float val = c[nt][v] + bias[n];
      const int which = n >> 10;        // 0=q 1=k 2=v
      const int rem = n & (DD - 1);
      const int h = rem >> 6, d = rem & 63;
      const int b = m >> 11, s = m & (SS - 1);
      const int bh = b * HH + h;
      const unsigned short bv = f2bf(val);
      if (which == 0)      Qb[((size_t)bh * SS + s) * HD + d] = bv;
      else if (which == 1) Kb[((size_t)bh * SS + s) * HD + d] = bv;
      else                 Vt[((size_t)bh * HD + d) * SS + s] = bv;
    }
  }
}

// ---------------------------------------------------------------------------
// Flash attention, one wave per 16-query block; keys streamed 32 at a time.
__global__ void attention_kernel(const unsigned short* __restrict__ Q,
                                 const unsigned short* __restrict__ Kb,
                                 const unsigned short* __restrict__ Vt,
                                 unsigned short* __restrict__ attn) {
  __shared__ unsigned short Plds[8][16 * 32];   // per-wave P staging (bf16)
  const int lane = threadIdx.x & 31;
  const int wave = threadIdx.x >> 5;
  const int g    = blockIdx.x * 8 + wave;
  const int bh   = g >> 7;                      // / (S/16)
  const int s0   = (g & 127) * 16;
  const int half = lane >> 4, col = lane & 15;

  // Preload Q A-fragments (K = HD = 64 -> two 16x32 frags), kept in VGPRs.
  const unsigned short* qbase = Q + ((size_t)bh * SS + s0) * HD;
  const Frag qa0 = load_a_frag(qbase, HD, lane);
  const Frag qa1 = load_a_frag(qbase + 32, HD, lane);

  v8f acc[4] = {};                              // 16 x 64 output accumulator
  float mrow[8], lrow[8];
#pragma unroll
  for (int v = 0; v < 8; ++v) { mrow[v] = -3.0e38f; lrow[v] = 0.0f; }

  unsigned short* Pw = Plds[wave];
  const int nkb = (s0 + 16 + 31) >> 5;          // causal: 32-key blocks needed

  for (int kbk = 0; kbk < nkb; ++kbk) {
    const int kBase = kbk * 32;

    // ---- scores: two 16x16 tiles, each = two K=32 WMMA steps over HD ----
    float st[2][8];
#pragma unroll
    for (int t = 0; t < 2; ++t) {
      const unsigned short* kb0 = Kb + ((size_t)bh * SS + kBase + t * 16) * HD;
      const Frag b0 = load_b_frag(kb0, HD, lane);
      const Frag b1 = load_b_frag(kb0 + 32, HD, lane);
      v8f s = {};
      s = wmma_bf16(qa0, b0, s);
      s = wmma_bf16(qa1, b1, s);
      const int kcol = kBase + t * 16 + col;
#pragma unroll
      for (int v = 0; v < 8; ++v) {
        const int qrow = s0 + v + 8 * half;
        float x = s[v] * 0.125f;                // 1/sqrt(64)
        if (kcol > qrow) x = -1.0e9f;           // causal mask (matches ref)
        st[t][v] = x;
      }
    }

    // ---- online softmax: row max across 16 lanes of each half ----
    float bm[8];
#pragma unroll
    for (int v = 0; v < 8; ++v) bm[v] = fmaxf(st[0][v], st[1][v]);
#pragma unroll
    for (int off = 1; off < 16; off <<= 1)
#pragma unroll
      for (int v = 0; v < 8; ++v) bm[v] = fmaxf(bm[v], __shfl_xor(bm[v], off, 32));

    float scale[8], psum[8];
#pragma unroll
    for (int v = 0; v < 8; ++v) {
      const float mn = fmaxf(mrow[v], bm[v]);
      scale[v] = exp2f((mrow[v] - mn) * LOG2E);
      mrow[v] = mn;
      psum[v] = 0.0f;
    }
#pragma unroll
    for (int nt = 0; nt < 4; ++nt)
#pragma unroll
      for (int v = 0; v < 8; ++v) acc[nt][v] *= scale[v];

    // ---- P = exp(s - m), bf16, staged to LDS in row-major 16x32 ----
#pragma unroll
    for (int t = 0; t < 2; ++t)
#pragma unroll
      for (int v = 0; v < 8; ++v) {
        const float p = exp2f((st[t][v] - mrow[v]) * LOG2E);
        psum[v] += p;
        Pw[(v + 8 * half) * 32 + t * 16 + col] = f2bf(p);
      }
#pragma unroll
    for (int off = 1; off < 16; off <<= 1)
#pragma unroll
      for (int v = 0; v < 8; ++v) psum[v] += __shfl_xor(psum[v], off, 32);
#pragma unroll
    for (int v = 0; v < 8; ++v) lrow[v] = lrow[v] * scale[v] + psum[v];

    // Wave-local LDS RAW: lanes wrote, lanes now read a different layout.
    asm volatile("s_wait_dscnt 0" ::: "memory");

    // ---- out += P(16x32) @ V(32x64): reload P as an A-fragment ----
    const Frag pa = load_a_frag(Pw, 32, lane);
#pragma unroll
    for (int nt = 0; nt < 4; ++nt) {
      const Frag vb =
          load_b_frag(Vt + ((size_t)bh * HD + nt * 16) * SS + kBase, SS, lane);
      acc[nt] = wmma_bf16(pa, vb, acc[nt]);
    }
  }

  // ---- normalize and store bf16 into attn [B, S, H*HD] ----
  const int b = bh >> 4, h = bh & 15;
#pragma unroll
  for (int v = 0; v < 8; ++v) {
    const float inv = 1.0f / lrow[v];
    const int s = s0 + v + 8 * half;
    unsigned short* orow = attn + ((size_t)b * SS + s) * DD + h * HD;
#pragma unroll
    for (int nt = 0; nt < 4; ++nt)
      orow[nt * 16 + col] = f2bf(acc[nt][v] * inv);
  }
}

// ---------------------------------------------------------------------------
// Output projection: [8192,1024]bf16 @ [1024,1024] + bias -> f32 out.
// Same TDM-fed 128x64 block tiling as the QKV GEMM.
__global__ void proj_gemm_kernel(const unsigned short* __restrict__ attn,
                                 const unsigned short* __restrict__ wT,
                                 const float* __restrict__ bias,
                                 float* __restrict__ out) {
  __shared__ unsigned short Btile[2][64 * 32];
  const int lane = threadIdx.x & 31;
  const int wave = threadIdx.x >> 5;
  const int tilesN = DD / 64;                    // 16
  const int m0 = (blockIdx.x / tilesN) * 128 + wave * 16;
  const int n0 = (blockIdx.x % tilesN) * 64;

  const unsigned long long wbase = (unsigned long long)(const void*)wT;
  const unsigned ldsb0 = (unsigned)(size_t)&Btile[0][0];
  const unsigned ldsb1 = (unsigned)(size_t)&Btile[1][0];

  if (wave == 0) {
    tdm_load_2d(ldsb0, wbase + (size_t)n0 * DD * 2, DD, DD, 32, 64, DD);
    __builtin_amdgcn_s_wait_tensorcnt(0);
  }
  __syncthreads();

  v8f c[4] = {};
  int cur = 0;
  for (int k0 = 0; k0 < DD; k0 += 32) {
    if (wave == 0 && k0 + 32 < DD)
      tdm_load_2d(cur ? ldsb0 : ldsb1,
                  wbase + ((size_t)n0 * DD + k0 + 32) * 2, DD, DD, 32, 64, DD);

    const Frag a = load_a_frag(attn + (size_t)m0 * DD + k0, DD, lane);
#pragma unroll
    for (int nt = 0; nt < 4; ++nt) {
      const Frag b = load_b_frag(&Btile[cur][nt * 16 * 32], 32, lane);
      c[nt] = wmma_bf16(a, b, c[nt]);
    }

    if (wave == 0 && k0 + 32 < DD) __builtin_amdgcn_s_wait_tensorcnt(0);
    __syncthreads();
    cur ^= 1;
  }

  const int half = lane >> 4, nc = lane & 15;
#pragma unroll
  for (int nt = 0; nt < 4; ++nt) {
#pragma unroll
    for (int v = 0; v < 8; ++v) {
      const int n = n0 + nt * 16 + nc;
      const int m = m0 + v + 8 * half;
      out[(size_t)m * DD + n] = c[nt][v] + bias[n];
    }
  }
}

// ---------------------------------------------------------------------------
extern "C" void kernel_launch(void* const* d_in, const int* in_sizes, int n_in,
                              void* d_out, int out_size, void* d_ws, size_t ws_size,
                              hipStream_t stream) {
  (void)in_sizes; (void)n_in; (void)out_size; (void)ws_size;
  const float* x      = (const float*)d_in[0];
  const float* w_qkv  = (const float*)d_in[1];
  const float* b_qkv  = (const float*)d_in[2];
  const float* w_proj = (const float*)d_in[3];
  const float* b_proj = (const float*)d_in[4];
  float* out = (float*)d_out;

  // Workspace layout (bf16 scratch, ~88 MB total), 256B aligned chunks.
  char* p = (char*)d_ws;
  auto take = [&](size_t bytes) {
    char* r = p;
    p += (bytes + 255) & ~(size_t)255;
    return r;
  };
  unsigned short* xb     = (unsigned short*)take((size_t)BB * SS * DD * 2);
  unsigned short* wqkvT  = (unsigned short*)take((size_t)N3 * DD * 2);
  unsigned short* wprojT = (unsigned short*)take((size_t)DD * DD * 2);
  unsigned short* Qb     = (unsigned short*)take((size_t)BH * SS * HD * 2);
  unsigned short* Kb     = (unsigned short*)take((size_t)BH * SS * HD * 2);
  unsigned short* Vt     = (unsigned short*)take((size_t)BH * HD * SS * 2);
  unsigned short* attn   = (unsigned short*)take((size_t)BB * SS * DD * 2);

  cvt_bf16_kernel<<<(BB * SS * DD) / (256 * 4), 256, 0, stream>>>(x, xb, BB * SS * DD);
  transpose_cvt_kernel<<<(DD * N3) / 256, 256, 0, stream>>>(w_qkv, wqkvT, DD, N3);
  transpose_cvt_kernel<<<(DD * DD) / 256, 256, 0, stream>>>(w_proj, wprojT, DD, DD);

  // QKV: 64 M-blocks (128 rows) x 48 N-blocks (64 cols) = 3072 workgroups.
  qkv_gemm_kernel<<<3072, 256, 0, stream>>>(xb, wqkvT, b_qkv, Qb, Kb, Vt);
  // Attention: 64 heads x 128 query blocks = 8192 waves / 8 per block.
  attention_kernel<<<1024, 256, 0, stream>>>(Qb, Kb, Vt, attn);
  // Proj: 64 M-blocks x 16 N-blocks = 1024 workgroups.
  proj_gemm_kernel<<<1024, 256, 0, stream>>>(attn, wprojT, b_proj, out);
}
